// GNN_76579266888290
// MI455X (gfx1250) — matrix operand order repaired and checked by reading, not compiled, not run on hardware
//
#include <hip/hip_runtime.h>

#define NN 40000
#define EE 640000
#define GG 2048
#define DD 128
#define TT 10
#define LL 5

typedef __attribute__((ext_vector_type(16))) _Float16 v16h;
typedef __attribute__((ext_vector_type(8)))  float    v8f;

// ---------------------------------------------------------------- utilities
__global__ void k_zero(float* p, int n) {
    int i = blockIdx.x * blockDim.x + threadIdx.x;
    if (i < n) p[i] = 0.0f;
}

// Fold BN into scale/shift and pre-transpose weights to f16 [Nout][K] rows so
// WMMA B-fragments are contiguous 32B loads.
__global__ void k_prep(const float* W1, const float* W2,
                       const float* b1, const float* g1, const float* bt1,
                       const float* m1, const float* v1,
                       const float* b2, const float* gO, const float* btO,
                       const float* mO, const float* vO,
                       _Float16* W1t, _Float16* W2t,
                       float* c1s, float* c1h, float* c2s, float* c2h) {
    int t = blockIdx.x * blockDim.x + threadIdx.x;
    const int total = LL * 256 * 128;
    if (t < total) {
        int l = t / (256 * 128);
        int rem = t % (256 * 128);
        {   // W1: [L,128,256] -> W1t: [L,256,128]  (row n holds column n of W1)
            int n = rem / 128, k = rem % 128;
            W1t[t] = (_Float16)W1[(size_t)(l * 128 + k) * 256 + n];
        }
        {   // W2: [L,256,128] -> W2t: [L,128,256]
            int n = rem / 256, k = rem % 256;
            W2t[t] = (_Float16)W2[(size_t)(l * 256 + k) * 128 + n];
        }
    }
    if (t < LL * 256) {
        float s = g1[t] * rsqrtf(v1[t] + 1e-5f);
        c1s[t] = s;
        c1h[t] = (b1[t] - m1[t]) * s + bt1[t];
    }
    if (t < LL * 128) {
        float s = gO[t] * rsqrtf(vO[t] + 1e-5f);
        c2s[t] = s;
        c2h[t] = (b2[t] - mO[t]) * s + btO[t];
    }
}

// ------------------------------------------------------------- atom encoder
// one wave32 per node; lane l handles cols 4l..4l+3
__global__ void k_atom(const int* __restrict__ xa, const float* __restrict__ aemb,
                       float* __restrict__ x) {
    int w    = (blockIdx.x * blockDim.x + threadIdx.x) >> 5;
    int lane = threadIdx.x & 31;
    if (w >= NN) return;
    float4 acc = make_float4(0.f, 0.f, 0.f, 0.f);
#pragma unroll
    for (int f = 0; f < 9; ++f) {
        int idx = xa[w * 9 + f];
        float4 r = ((const float4*)(aemb + (size_t)(f * 64 + idx) * DD))[lane];
        acc.x += r.x; acc.y += r.y; acc.z += r.z; acc.w += r.w;
    }
    ((float4*)(x + (size_t)w * DD))[lane] = acc;
}

// ------------------------------------------------------------ edge messages
// one wave32 per edge: e = sum bond embeddings, msg = relu(x[src]+e),
// scatter-add into agg[dst]. x/agg are L2-resident (20.5MB each, 192MB L2).
__global__ void k_edge(const int* __restrict__ ei, const int* __restrict__ ea,
                       const float* __restrict__ bemb, const float* __restrict__ x,
                       float* __restrict__ agg, int layer) {
    int e    = (blockIdx.x * blockDim.x + threadIdx.x) >> 5;
    int lane = threadIdx.x & 31;
    if (e >= EE) return;
    int s = ei[e];
    int d = ei[EE + e];
    const float* base = bemb + (size_t)layer * 3 * 8 * DD;
    float4 acc = ((const float4*)(x + (size_t)s * DD))[lane];
#pragma unroll
    for (int f = 0; f < 3; ++f) {
        int a = ea[e * 3 + f];
        float4 r = ((const float4*)(base + (size_t)(f * 8 + a) * DD))[lane];
        acc.x += r.x; acc.y += r.y; acc.z += r.z; acc.w += r.w;
    }
    acc.x = acc.x > 0.f ? acc.x : 0.f;
    acc.y = acc.y > 0.f ? acc.y : 0.f;
    acc.z = acc.z > 0.f ? acc.z : 0.f;
    acc.w = acc.w > 0.f ? acc.w : 0.f;
    float* ap = agg + (size_t)d * DD + lane * 4;
    atomicAdd(ap + 0, acc.x);
    atomicAdd(ap + 1, acc.y);
    atomicAdd(ap + 2, acc.z);
    atomicAdd(ap + 3, acc.w);
}

// ------------------------------------------------------ fused MLP via WMMA
// 64-node tile per block (512 thr = 16 waves). h0=(1+eps)x+agg staged as f16
// in LDS; GEMM1(128->256)+BN+ReLU -> LDS f16; GEMM2(256->128)+BN(+ReLU) -> x.
// Per-wave the output column-block (nt) is fixed, so all B-fragments are
// hoisted into registers and reused across the m-tiles.
// A-frag: lane m=l&15, grp=l>>4, elem j -> k =
// grp*16+j.  B-frag: lane n=l&15, elem j -> k = grp*16+j,
// contiguous in the pre-transposed weight row.
__global__ __launch_bounds__(512) void k_mlp(
        float* __restrict__ x, const float* __restrict__ agg,
        const float* __restrict__ epsArr,
        const _Float16* __restrict__ W1t, const _Float16* __restrict__ W2t,
        const float* __restrict__ c1s, const float* __restrict__ c1h,
        const float* __restrict__ c2s, const float* __restrict__ c2h,
        int layer) {
    __shared__ _Float16 sH[64][136];   // h0 tile, f16, padded
    __shared__ _Float16 sT[64][264];   // GEMM1 output tile, f16, padded

    const int tid  = threadIdx.x;
    const int base = blockIdx.x * 64;
    const float e1 = 1.0f + epsArr[layer];

    // stage h0 = (1+eps)*x + agg  (16 elems per thread)
    {
        int row = tid >> 3;
        int c0  = (tid & 7) * 16;
        const float* xr = x   + (size_t)(base + row) * DD + c0;
        const float* ar = agg + (size_t)(base + row) * DD + c0;
#pragma unroll
        for (int j = 0; j < 16; ++j)
            sH[row][c0 + j] = (_Float16)(e1 * xr[j] + ar[j]);
    }
    __syncthreads();

    const int wave = tid >> 5;
    const int lane = tid & 31;
    const int m16  = lane & 15;
    const int grp  = lane >> 4;

    // ---- GEMM1: [64x128] @ [128x256] -> BN+ReLU -> sT (f16)
    {
        const _Float16* W1l = W1t + (size_t)layer * 256 * 128;
        const int nt = wave;                       // fixed column-block
        const int n  = nt * 16 + m16;
        const _Float16* wrow = W1l + (size_t)n * 128;
        v16h B[4];
#pragma unroll
        for (int ks = 0; ks < 4; ++ks)
            B[ks] = *(const v16h*)(wrow + ks * 32 + grp * 16);
        const float s  = c1s[layer * 256 + n];
        const float sh = c1h[layer * 256 + n];
#pragma unroll
        for (int mt = 0; mt < 4; ++mt) {
            v8f acc = {};
            const _Float16* hrow = &sH[mt * 16 + m16][0];
#pragma unroll
            for (int ks = 0; ks < 4; ++ks) {
                v16h a;
#pragma unroll
                for (int v = 0; v < 8; ++v) {
                    int kp = ks * 32 + grp * 16 + v * 2;
                    a[2 * v]     = hrow[kp];
                    a[2 * v + 1] = hrow[kp + 1];
                }
                acc = __builtin_amdgcn_wmma_f32_16x16x32_f16(
                          false, a, false, B[ks], (short)0, acc, false, false);
            }
#pragma unroll
            for (int r = 0; r < 8; ++r) {
                float vv = acc[r] * s + sh;
                vv = vv > 0.f ? vv : 0.f;
                sT[mt * 16 + grp * 8 + r][n] = (_Float16)vv;
            }
        }
    }
    __syncthreads();

    // ---- GEMM2: [64x256] @ [256x128] -> BN (+ReLU) -> x (in place)
    {
        const _Float16* W2l = W2t + (size_t)layer * 128 * 256;
        const int nt = wave & 7;                   // fixed column-block
        const int n  = nt * 16 + m16;
        const _Float16* wrow = W2l + (size_t)n * 256;
        v16h B[8];
#pragma unroll
        for (int ks = 0; ks < 8; ++ks)
            B[ks] = *(const v16h*)(wrow + ks * 32 + grp * 16);
        const float s  = c2s[layer * 128 + n];
        const float sh = c2h[layer * 128 + n];
        const bool dorelu = (layer < LL - 1);
#pragma unroll
        for (int t = 0; t < 2; ++t) {
            int mt = (wave >> 3) + 2 * t;          // 2 m-tiles per wave
            v8f acc = {};
            const _Float16* trow = &sT[mt * 16 + m16][0];
#pragma unroll
            for (int ks = 0; ks < 8; ++ks) {
                v16h a;
#pragma unroll
                for (int v = 0; v < 8; ++v) {
                    int kp = ks * 32 + grp * 16 + v * 2;
                    a[2 * v]     = trow[kp];
                    a[2 * v + 1] = trow[kp + 1];
                }
                acc = __builtin_amdgcn_wmma_f32_16x16x32_f16(
                          false, a, false, B[ks], (short)0, acc, false, false);
            }
#pragma unroll
            for (int r = 0; r < 8; ++r) {
                float vv = acc[r] * s + sh;
                if (dorelu) vv = vv > 0.f ? vv : 0.f;
                x[(size_t)(base + mt * 16 + grp * 8 + r) * DD + n] = vv;
            }
        }
    }
}

// ---------------------------------------------------------------- pooling
__global__ void k_pool(const float* __restrict__ x, const int* __restrict__ batch,
                       float* __restrict__ sums, float* __restrict__ cnt) {
    int w    = (blockIdx.x * blockDim.x + threadIdx.x) >> 5;
    int lane = threadIdx.x & 31;
    if (w >= NN) return;
    int g = batch[w];
    float4 v = ((const float4*)(x + (size_t)w * DD))[lane];
    float* sp = sums + (size_t)g * DD + lane * 4;
    atomicAdd(sp + 0, v.x);
    atomicAdd(sp + 1, v.y);
    atomicAdd(sp + 2, v.z);
    atomicAdd(sp + 3, v.w);
    if (lane == 0) atomicAdd(cnt + g, 1.0f);
}

__global__ void k_head(const float* __restrict__ sums, const float* __restrict__ cnt,
                       const float* __restrict__ Wp, const float* __restrict__ bp,
                       float* __restrict__ out) {
    int i = blockIdx.x * blockDim.x + threadIdx.x;
    if (i >= GG * TT) return;
    int g = i / TT, t = i % TT;
    float c = cnt[g];
    c = c > 1.f ? c : 1.f;
    float inv = 1.f / c;
    float acc = bp[t];
    const float* sr = sums + (size_t)g * DD;
    for (int k = 0; k < DD; ++k) acc += sr[k] * inv * Wp[k * TT + t];
    out[i] = acc;
}

// ------------------------------------------------------------------ driver
extern "C" void kernel_launch(void* const* d_in, const int* in_sizes, int n_in,
                              void* d_out, int out_size, void* d_ws, size_t ws_size,
                              hipStream_t stream) {
    (void)in_sizes; (void)n_in; (void)out_size; (void)ws_size;
    const int*   x_atom = (const int*)d_in[0];
    const int*   ei     = (const int*)d_in[1];
    const int*   ea     = (const int*)d_in[2];
    const int*   batch  = (const int*)d_in[3];
    const float* aemb   = (const float*)d_in[4];
    const float* bemb   = (const float*)d_in[5];
    const float* eps    = (const float*)d_in[6];
    const float* W1     = (const float*)d_in[7];
    const float* b1     = (const float*)d_in[8];
    const float* g1     = (const float*)d_in[9];
    const float* bt1    = (const float*)d_in[10];
    const float* m1     = (const float*)d_in[11];
    const float* v1     = (const float*)d_in[12];
    const float* W2     = (const float*)d_in[13];
    const float* b2     = (const float*)d_in[14];
    const float* gO     = (const float*)d_in[15];
    const float* btO    = (const float*)d_in[16];
    const float* mO     = (const float*)d_in[17];
    const float* vO     = (const float*)d_in[18];
    const float* Wp     = (const float*)d_in[19];
    const float* bp     = (const float*)d_in[20];
    float* out = (float*)d_out;

    char* ws = (char*)d_ws;
    size_t off = 0;
    float*    x    = (float*)(ws + off);    off += (size_t)NN * DD * 4;       // 20.48 MB
    float*    agg  = (float*)(ws + off);    off += (size_t)NN * DD * 4;       // 20.48 MB
    _Float16* W1t  = (_Float16*)(ws + off); off += (size_t)LL * 256 * 128 * 2;
    _Float16* W2t  = (_Float16*)(ws + off); off += (size_t)LL * 128 * 256 * 2;
    float*    c1s  = (float*)(ws + off);    off += LL * 256 * 4;
    float*    c1h  = (float*)(ws + off);    off += LL * 256 * 4;
    float*    c2s  = (float*)(ws + off);    off += LL * 128 * 4;
    float*    c2h  = (float*)(ws + off);    off += LL * 128 * 4;
    float*    sums = (float*)(ws + off);    off += (size_t)GG * DD * 4;       // 1 MB
    float*    cnt  = (float*)(ws + off);    off += GG * 4;                    // cnt follows sums

    k_prep<<<(LL * 256 * 128 + 255) / 256, 256, 0, stream>>>(
        W1, W2, b1, g1, bt1, m1, v1, b2, gO, btO, mO, vO,
        W1t, W2t, c1s, c1h, c2s, c2h);

    k_atom<<<(NN * 32 + 255) / 256, 256, 0, stream>>>(x_atom, aemb, x);

    for (int l = 0; l < LL; ++l) {
        k_zero<<<(NN * DD + 255) / 256, 256, 0, stream>>>(agg, NN * DD);
        k_edge<<<(EE * 32 + 255) / 256, 256, 0, stream>>>(ei, ea, bemb, x, agg, l);
        k_mlp<<<NN / 64, 512, 0, stream>>>(x, agg, eps, W1t, W2t,
                                           c1s, c1h, c2s, c2h, l);
    }

    k_zero<<<(GG * DD + GG + 255) / 256, 256, 0, stream>>>(sums, GG * DD + GG);
    k_pool<<<(NN * 32 + 255) / 256, 256, 0, stream>>>(x, batch, sums, cnt);
    k_head<<<(GG * TT + 255) / 256, 256, 0, stream>>>(sums, cnt, Wp, bp, out);
}
